// RITS_10840497455545
// MI455X (gfx1250) — compile-verified, matrix operand order checked by reading, not run on hardware
//
#include <hip/hip_runtime.h>
#include <hip/hip_bf16.h>

typedef __attribute__((ext_vector_type(16))) __bf16       v16bf;
typedef __attribute__((ext_vector_type(8)))  float        v8f;
typedef __attribute__((ext_vector_type(8)))  unsigned int v8u;

#define T_SEQ 100
#define B_SZ  1024
#define F_SZ  64
#define H_SZ  256
#define ROWS  16
#define NWG   (B_SZ / ROWS)   // 64 workgroups
#define NTHR  256             // 8 waves of 32

// ---------- helpers ----------
__device__ __forceinline__ unsigned short f2bf(float f) {
  unsigned int u = __builtin_bit_cast(unsigned int, f);
  u += 0x7FFFu + ((u >> 16) & 1u);          // round-to-nearest-even
  return (unsigned short)(u >> 16);
}
__device__ __forceinline__ float sigmoidf_(float x) { return 1.0f / (1.0f + __expf(-x)); }

__device__ __forceinline__ v16bf as_bf(v8u u) {
  union { v8u u8; v16bf b16; } c; c.u8 = u; return c.b16;
}
__device__ __forceinline__ v8f vzero8() {
  v8f z = {0.f,0.f,0.f,0.f,0.f,0.f,0.f,0.f}; return z;
}

// Fragment loader for 16x32 bf16 tiles, row-major [row][k] source.
// Works for both A (act rows) and B (weight rows == W^T cols) operands.
// lane L: row = L&15 ; k = 2*(v%4) + 16*(v/4) + 8*(L>>4)
// (backend merges the 8 b32 reads into two b128 loads)
__device__ __forceinline__ v8u load_frag(const unsigned short* base, int ld, int lane) {
  const int r = lane & 15;
  const unsigned short* p = base + r * ld + ((lane >> 4) << 3);
  v8u u;
#pragma unroll
  for (int v = 0; v < 8; ++v) {
    const int k = ((v & 3) << 1) + ((v >> 2) << 4);
    u[v] = *reinterpret_cast<const unsigned int*>(p + k);
  }
  return u;
}

__device__ __forceinline__ v8f wmma_bf16(v8u a, v8u b, v8f c) {
  return __builtin_amdgcn_wmma_f32_16x16x32_bf16(
      false, as_bf(a), false, as_bf(b), (short)0, c, false, false);
}

// ---------- LDS budget (bytes) ----------
static constexpr int SMEM_BYTES =
    (ROWS*H_SZ*4)*2        // s_hf, s_cf
  + (ROWS*4*H_SZ*4)        // s_gates
  + (ROWS*F_SZ*4)*5        // s_x, s_m, s_xh, s_zh, s_al
  + (H_SZ*4)               // s_bdh
  + (F_SZ*4)*3             // s_bhist, s_bfeat, s_bcomb
  + (4*H_SZ*4)             // s_bihh
  + (F_SZ*4)*2             // s_bdx, s_wdx
  + 128                    // s_wred
  + (ROWS*H_SZ*2)          // s_hb
  + (ROWS*F_SZ*2)*2        // s_db, s_xcb
  + (ROWS*2*F_SZ*2)*2      // s_gm, s_in2
  + (H_SZ*F_SZ*2)*2        // s_Wdh, s_Whist
  + (F_SZ*F_SZ*2)          // s_Wfeat
  + (F_SZ*2*F_SZ*2);       // s_Wcomb   => 235,904 B < 320 KB

// ---------- kernel 1: convert weights to bf16 (W_feat diag-masked) ----------
__global__ void rits_prep_weights(
    const float* __restrict__ Wdh, const float* __restrict__ Whist,
    const float* __restrict__ Wfeat, const float* __restrict__ Wcomb,
    const float* __restrict__ Wih, const float* __restrict__ Whh,
    unsigned short* __restrict__ oWdh, unsigned short* __restrict__ oWhist,
    unsigned short* __restrict__ oWfeat, unsigned short* __restrict__ oWcomb,
    unsigned short* __restrict__ oWih, unsigned short* __restrict__ oWhh) {
  const int id = blockIdx.x * blockDim.x + threadIdx.x;
  const int st = gridDim.x * blockDim.x;
  for (int i = id; i < H_SZ*F_SZ;     i += st) oWdh[i]   = f2bf(Wdh[i]);
  for (int i = id; i < F_SZ*H_SZ;     i += st) oWhist[i] = f2bf(Whist[i]);
  for (int i = id; i < F_SZ*F_SZ;     i += st) {
    int r = i >> 6, c = i & 63;
    oWfeat[i] = f2bf(r == c ? 0.f : Wfeat[i]);
  }
  for (int i = id; i < F_SZ*2*F_SZ;   i += st) oWcomb[i] = f2bf(Wcomb[i]);
  for (int i = id; i < 4*H_SZ*2*F_SZ; i += st) oWih[i]   = f2bf(Wih[i]);
  for (int i = id; i < 4*H_SZ*H_SZ;   i += st) oWhh[i]   = f2bf(Whh[i]);
}

// ---------- kernel 2: per-step mask denominators ----------
__global__ void rits_mask_sums(const float* __restrict__ Mk, float* __restrict__ msum) {
  const int t = blockIdx.x;
  float s = 0.f;
  for (int i = threadIdx.x; i < B_SZ * F_SZ; i += blockDim.x) {
    int b = i >> 6, f = i & 63;
    s += Mk[((size_t)b * T_SEQ + t) * F_SZ + f];
  }
#pragma unroll
  for (int off = 16; off > 0; off >>= 1) s += __shfl_down(s, off, 32);
  __shared__ float red[8];
  int lane = threadIdx.x & 31, w = threadIdx.x >> 5;
  if (lane == 0) red[w] = s;
  __syncthreads();
  if (threadIdx.x == 0) {
    float tt = 0.f;
    for (int i = 0; i < (int)(blockDim.x >> 5); ++i) tt += red[i];
    msum[t] = tt;
  }
}

// ---------- kernel 3: persistent recurrent kernel ----------
__global__ __launch_bounds__(NTHR, 1)
void rits_persistent(
    const float* __restrict__ X, const float* __restrict__ Mk, const float* __restrict__ Dl,
    const unsigned short* __restrict__ gWdh, const unsigned short* __restrict__ gWhist,
    const unsigned short* __restrict__ gWfeat, const unsigned short* __restrict__ gWcomb,
    const unsigned short* __restrict__ gWih, const unsigned short* __restrict__ gWhh,
    const float* __restrict__ b_dh, const float* __restrict__ W_dx, const float* __restrict__ b_dx,
    const float* __restrict__ b_hist, const float* __restrict__ b_feat, const float* __restrict__ b_comb,
    const float* __restrict__ b_ih, const float* __restrict__ b_hh,
    float* __restrict__ imputed, float* __restrict__ partials) {
  extern __shared__ __align__(16) char smem_raw[];
  char* p = smem_raw;
  float* s_hf    = (float*)p;          p += ROWS*H_SZ*4;
  float* s_cf    = (float*)p;          p += ROWS*H_SZ*4;
  float* s_gates = (float*)p;          p += ROWS*4*H_SZ*4;
  float* s_x     = (float*)p;          p += ROWS*F_SZ*4;
  float* s_m     = (float*)p;          p += ROWS*F_SZ*4;
  float* s_xh    = (float*)p;          p += ROWS*F_SZ*4;
  float* s_zh    = (float*)p;          p += ROWS*F_SZ*4;
  float* s_al    = (float*)p;          p += ROWS*F_SZ*4;
  float* s_bdh   = (float*)p;          p += H_SZ*4;
  float* s_bhist = (float*)p;          p += F_SZ*4;
  float* s_bfeat = (float*)p;          p += F_SZ*4;
  float* s_bcomb = (float*)p;          p += F_SZ*4;
  float* s_bihh  = (float*)p;          p += 4*H_SZ*4;
  float* s_bdx   = (float*)p;          p += F_SZ*4;
  float* s_wdx   = (float*)p;          p += F_SZ*4;
  float* s_wred  = (float*)p;          p += 128;
  unsigned short* s_hb    = (unsigned short*)p; p += ROWS*H_SZ*2;
  unsigned short* s_db    = (unsigned short*)p; p += ROWS*F_SZ*2;
  unsigned short* s_xcb   = (unsigned short*)p; p += ROWS*F_SZ*2;
  unsigned short* s_gm    = (unsigned short*)p; p += ROWS*2*F_SZ*2;
  unsigned short* s_in2   = (unsigned short*)p; p += ROWS*2*F_SZ*2;
  unsigned short* s_Wdh   = (unsigned short*)p; p += H_SZ*F_SZ*2;
  unsigned short* s_Whist = (unsigned short*)p; p += F_SZ*H_SZ*2;
  unsigned short* s_Wfeat = (unsigned short*)p; p += F_SZ*F_SZ*2;
  unsigned short* s_Wcomb = (unsigned short*)p; p += F_SZ*2*F_SZ*2;

  const int tid  = threadIdx.x;
  const int lane = tid & 31;
  const int wave = tid >> 5;
  const int row0 = blockIdx.x * ROWS;

  // one-time staging: small weights + biases into LDS, zero state
  for (int i = tid; i < H_SZ*F_SZ;   i += NTHR) s_Wdh[i]   = gWdh[i];
  for (int i = tid; i < F_SZ*H_SZ;   i += NTHR) s_Whist[i] = gWhist[i];
  for (int i = tid; i < F_SZ*F_SZ;   i += NTHR) s_Wfeat[i] = gWfeat[i];
  for (int i = tid; i < F_SZ*2*F_SZ; i += NTHR) s_Wcomb[i] = gWcomb[i];
  for (int i = tid; i < H_SZ;        i += NTHR) s_bdh[i]   = b_dh[i];
  for (int i = tid; i < F_SZ;        i += NTHR) {
    s_bhist[i] = b_hist[i]; s_bfeat[i] = b_feat[i]; s_bcomb[i] = b_comb[i];
    s_bdx[i]   = b_dx[i];   s_wdx[i]   = W_dx[i * F_SZ + i];   // diag
  }
  for (int i = tid; i < 4*H_SZ;      i += NTHR) s_bihh[i] = b_ih[i] + b_hh[i];
  for (int i = tid; i < ROWS*H_SZ;   i += NTHR) { s_hf[i] = 0.f; s_cf[i] = 0.f; s_hb[i] = 0; }
  __syncthreads();

  for (int t = 0; t < T_SEQ; ++t) {
    float sa = 0.f, sb = 0.f, se = 0.f;

    // ---- phase 0: load x,m,d tile; elementwise gamma_x; build [gamma_x|m] bf16 ----
    for (int i = tid; i < ROWS*F_SZ; i += NTHR) {
      int r = i >> 6, f = i & 63;
      size_t g = ((size_t)(row0 + r) * T_SEQ + t) * F_SZ + f;
      float xv = X[g], mv = Mk[g], dv = Dl[g];
      // prefetch next timestep's cachelines (row stride 25.6KB -> no reuse
      // without this); global_prefetch_b8, no counter cost
      if (t + 1 < T_SEQ && (f & 31) == 0) {
        __builtin_prefetch(&X[g + F_SZ], 0, 3);
        __builtin_prefetch(&Mk[g + F_SZ], 0, 3);
        __builtin_prefetch(&Dl[g + F_SZ], 0, 3);
      }
      s_x[i] = xv; s_m[i] = mv;
      s_db[i] = f2bf(dv);
      float gx = __expf(-fmaxf(dv * s_wdx[f] + s_bdx[f], 0.f));
      s_gm[r*128 + f]      = f2bf(gx);
      s_gm[r*128 + 64 + f] = f2bf(mv);
    }
    __syncthreads();

    // ---- phase 1: gamma_h = exp(-relu(d @ Wdh^T + b)); h *= gamma_h ----
#pragma unroll
    for (int tix = 0; tix < 2; ++tix) {
      int n0 = (wave * 2 + tix) * 16;
      v8f acc = vzero8();
#pragma unroll
      for (int kt = 0; kt < 2; ++kt)
        acc = wmma_bf16(load_frag(s_db + kt*32, F_SZ, lane),
                        load_frag(s_Wdh + n0*F_SZ + kt*32, F_SZ, lane), acc);
      int n = n0 + (lane & 15);
      int mb = (lane >> 4) << 3;
      float bn = s_bdh[n];
#pragma unroll
      for (int v = 0; v < 8; ++v) {
        float g = __expf(-fmaxf(acc[v] + bn, 0.f));
        int idx = (v + mb) * H_SZ + n;
        float hv = s_hf[idx] * g;
        s_hf[idx] = hv;
        s_hb[idx] = f2bf(hv);
      }
    }
    __syncthreads();

    // ---- phase 2: x_h = h @ Whist^T + b_hist  (waves 0-3) ----
    if (wave < 4) {
      int n0 = wave * 16;
      v8f acc = vzero8();
#pragma unroll
      for (int kt = 0; kt < 8; ++kt)
        acc = wmma_bf16(load_frag(s_hb + kt*32, H_SZ, lane),
                        load_frag(s_Whist + n0*H_SZ + kt*32, H_SZ, lane), acc);
      int n = n0 + (lane & 15);
      int mb = (lane >> 4) << 3;
      float bn = s_bhist[n];
#pragma unroll
      for (int v = 0; v < 8; ++v) s_xh[(v + mb) * F_SZ + n] = acc[v] + bn;
    }
    __syncthreads();

    // ---- phase 2b: loss a; x_c = m*x + (1-m)*x_h (bf16) ----
    for (int i = tid; i < ROWS*F_SZ; i += NTHR) {
      float xh = s_xh[i], xv = s_x[i], mv = s_m[i];
      sa += fabsf(xh - xv) * mv;
      s_xcb[i] = f2bf(mv * xv + (1.f - mv) * xh);
    }
    __syncthreads();

    // ---- phase 3: z_h (waves 0-3) || alpha-pre (waves 4-7) ----
    if (wave < 4) {
      int n0 = wave * 16;
      v8f acc = vzero8();
#pragma unroll
      for (int kt = 0; kt < 2; ++kt)
        acc = wmma_bf16(load_frag(s_xcb + kt*32, F_SZ, lane),
                        load_frag(s_Wfeat + n0*F_SZ + kt*32, F_SZ, lane), acc);
      int n = n0 + (lane & 15);
      int mb = (lane >> 4) << 3;
      float bn = s_bfeat[n];
#pragma unroll
      for (int v = 0; v < 8; ++v) s_zh[(v + mb) * F_SZ + n] = acc[v] + bn;
    } else {
      int n0 = (wave - 4) * 16;
      v8f acc = vzero8();
#pragma unroll
      for (int kt = 0; kt < 4; ++kt)
        acc = wmma_bf16(load_frag(s_gm + kt*32, 2*F_SZ, lane),
                        load_frag(s_Wcomb + n0*2*F_SZ + kt*32, 2*F_SZ, lane), acc);
      int n = n0 + (lane & 15);
      int mb = (lane >> 4) << 3;
      float bn = s_bcomb[n];
#pragma unroll
      for (int v = 0; v < 8; ++v) s_al[(v + mb) * F_SZ + n] = acc[v] + bn;
    }
    __syncthreads();

    // ---- phase 4: losses b,e; c_h; c_c -> imputed + LSTM input [c_c|m] ----
    for (int i = tid; i < ROWS*F_SZ; i += NTHR) {
      int r = i >> 6, f = i & 63;
      float al = sigmoidf_(s_al[i]);
      float zh = s_zh[i], xh = s_xh[i], xv = s_x[i], mv = s_m[i];
      sb += fabsf(zh - xv) * mv;
      float ch = al * zh + (1.f - al) * xh;
      se += fabsf(ch - xv) * mv;
      float ccv = mv * xv + (1.f - mv) * ch;
      imputed[((size_t)(row0 + r) * T_SEQ + t) * F_SZ + f] = ccv;
      s_in2[r*128 + f]      = f2bf(ccv);
      s_in2[r*128 + 64 + f] = f2bf(mv);
    }
    __syncthreads();

    // ---- phase 5: gates = [c_c,m]@Wih^T + h@Whh^T + (b_ih+b_hh) ----
    {
      v8u aI[4], aH[8];
#pragma unroll
      for (int kt = 0; kt < 4; ++kt) aI[kt] = load_frag(s_in2 + kt*32, 2*F_SZ, lane);
#pragma unroll
      for (int kt = 0; kt < 8; ++kt) aH[kt] = load_frag(s_hb + kt*32, H_SZ, lane);
      for (int nt = wave * 8; nt < wave * 8 + 8; ++nt) {
        int n0 = nt * 16;
        v8f acc = vzero8();
#pragma unroll
        for (int kt = 0; kt < 4; ++kt)
          acc = wmma_bf16(aI[kt], load_frag(gWih + n0*2*F_SZ + kt*32, 2*F_SZ, lane), acc);
#pragma unroll
        for (int kt = 0; kt < 8; ++kt)
          acc = wmma_bf16(aH[kt], load_frag(gWhh + n0*H_SZ + kt*32, H_SZ, lane), acc);
        int n = n0 + (lane & 15);
        int mb = (lane >> 4) << 3;
        float bn = s_bihh[n];
#pragma unroll
        for (int v = 0; v < 8; ++v) s_gates[(v + mb) * 4*H_SZ + n] = acc[v] + bn;
      }
    }
    __syncthreads();

    // ---- phase 6: LSTM cell elementwise ----
    for (int i = tid; i < ROWS*H_SZ; i += NTHR) {
      int r = i >> 8, j = i & 255;
      const float* gr = s_gates + r * 4*H_SZ;
      float ig = sigmoidf_(gr[j]);
      float fg = sigmoidf_(gr[H_SZ + j]);
      float gg = tanhf(gr[2*H_SZ + j]);
      float og = sigmoidf_(gr[3*H_SZ + j]);
      float cv = fg * s_cf[i] + ig * gg;
      s_cf[i] = cv;
      float hv = og * tanhf(cv);
      s_hf[i] = hv;
      s_hb[i] = f2bf(hv);
    }
    __syncthreads();

    // ---- phase 7: per-step loss partials (deterministic, no atomics) ----
#pragma unroll
    for (int off = 16; off > 0; off >>= 1) {
      sa += __shfl_down(sa, off, 32);
      sb += __shfl_down(sb, off, 32);
      se += __shfl_down(se, off, 32);
    }
    if (lane == 0) { s_wred[wave*3+0] = sa; s_wred[wave*3+1] = sb; s_wred[wave*3+2] = se; }
    __syncthreads();
    if (tid == 0) {
      float ta = 0.f, tb = 0.f, te = 0.f;
      for (int w = 0; w < 8; ++w) { ta += s_wred[w*3]; tb += s_wred[w*3+1]; te += s_wred[w*3+2]; }
      float* q = partials + ((size_t)blockIdx.x * T_SEQ + t) * 3;
      q[0] = ta; q[1] = tb; q[2] = te;
    }
    __syncthreads();
  }
}

// ---------- kernel 4: final scalar reduction (reference loss bookkeeping) ----------
__global__ void rits_finalize(const float* __restrict__ partials,
                              const float* __restrict__ msum,
                              float* __restrict__ out_scalars) {
  if (blockIdx.x == 0 && threadIdx.x == 0) {
    float loss = 0.f, mae = 0.f, cum = 0.f;
    for (int t = 0; t < T_SEQ; ++t) {
      float a = 0.f, b = 0.f, e = 0.f;
      for (int w = 0; w < NWG; ++w) {
        const float* q = partials + ((size_t)w * T_SEQ + t) * 3;
        a += q[0]; b += q[1]; e += q[2];
      }
      float dnm = msum[t] + 1e-9f;
      float an = a / dnm, bn = b / dnm, en = e / dnm;
      mae += en;                 // rec_mae accumulates e_t
      cum += en;                 // cumulative rec_mae
      loss += an + bn + cum;     // rec_loss += a + b + rec_mae(cumulative)
    }
    out_scalars[0] = mae / (float)T_SEQ;
    out_scalars[1] = loss / (float)(T_SEQ * 3);
  }
}

// ---------- launcher ----------
extern "C" void kernel_launch(void* const* d_in, const int* in_sizes, int n_in,
                              void* d_out, int out_size, void* d_ws, size_t ws_size,
                              hipStream_t stream) {
  const float* X      = (const float*)d_in[0];
  const float* Mk     = (const float*)d_in[1];
  const float* Dl     = (const float*)d_in[2];
  const float* W_dh   = (const float*)d_in[3];
  const float* b_dh   = (const float*)d_in[4];
  const float* W_dx   = (const float*)d_in[5];
  const float* b_dx   = (const float*)d_in[6];
  const float* W_hist = (const float*)d_in[7];
  const float* b_hist = (const float*)d_in[8];
  const float* W_feat = (const float*)d_in[9];
  const float* b_feat = (const float*)d_in[10];
  const float* W_comb = (const float*)d_in[11];
  const float* b_comb = (const float*)d_in[12];
  const float* W_ih   = (const float*)d_in[13];
  const float* b_ih   = (const float*)d_in[14];
  const float* W_hh   = (const float*)d_in[15];
  const float* b_hh   = (const float*)d_in[16];

  char* ws = (char*)d_ws;
  unsigned short* wdh   = (unsigned short*)(ws + 0);        // 32768
  unsigned short* whist = (unsigned short*)(ws + 32768);    // 32768
  unsigned short* wfeat = (unsigned short*)(ws + 65536);    // 8192
  unsigned short* wcomb = (unsigned short*)(ws + 73728);    // 16384
  unsigned short* wih   = (unsigned short*)(ws + 90112);    // 262144
  unsigned short* whh   = (unsigned short*)(ws + 352256);   // 524288
  float*          parts = (float*)(ws + 876544);            // 64*100*3*4 = 76800
  float*          msum  = (float*)(ws + 953344);            // 400

  float* imputed = (float*)d_out;
  float* scalars = imputed + (size_t)B_SZ * T_SEQ * F_SZ;   // indices 6553600, 6553601

  rits_prep_weights<<<512, 256, 0, stream>>>(W_dh, W_hist, W_feat, W_comb, W_ih, W_hh,
                                             wdh, whist, wfeat, wcomb, wih, whh);
  rits_mask_sums<<<T_SEQ, 256, 0, stream>>>(Mk, msum);
  rits_persistent<<<NWG, NTHR, SMEM_BYTES, stream>>>(
      X, Mk, Dl, wdh, whist, wfeat, wcomb, wih, whh,
      b_dh, W_dx, b_dx, b_hist, b_feat, b_comb, b_ih, b_hh,
      imputed, parts);
  rits_finalize<<<1, 32, 0, stream>>>(parts, msum, scalars);
}
